// QuantumInspiredMultiInterestEncoder_28965259444417
// MI455X (gfx1250) — matrix operand order repaired and checked by reading, not compiled, not run on hardware
//
#include <hip/hip_runtime.h>
#include <cstddef>

// ---------------------------------------------------------------------------
// MI455X / gfx1250 implementation.
// Precision: bf16 WMMA (v_wmma_f32_16x16x32_bf16) with f32 accumulate for all
// GEMMs (compute-bound; ~250 GFLOP vs ~400MB traffic => matrix-core bound).
// GEMM: 128x128 block tile, 8 waves (wave32) as 4Mx2N, wave tile 32x64
// (2x4 WMMA subtiles), K-step 32, double-buffered LDS fed by
// GLOBAL_LOAD_ASYNC_TO_LDS_B128 (ASYNCcnt) when the toolchain exposes it.
// ---------------------------------------------------------------------------

typedef __bf16 bf16_t;
typedef __attribute__((ext_vector_type(8)))  __bf16 v8bf;
typedef __attribute__((ext_vector_type(16))) __bf16 v16bf;
typedef __attribute__((ext_vector_type(8)))  float  v8f;
typedef __attribute__((ext_vector_type(4)))  int    v4i;
typedef __attribute__((address_space(1))) v4i* gv4i_p;   // global b128 operand
typedef __attribute__((address_space(3))) v4i* lv4i_p;   // LDS b128 operand

union FragBF { v16bf v; v8bf h[2]; };

#if __has_builtin(__builtin_amdgcn_global_load_async_to_lds_b128) && \
    __has_builtin(__builtin_amdgcn_s_wait_asynccnt)
#define GEMM_ASYNC 1
#else
#define GEMM_ASYNC 0
#endif

__device__ __forceinline__ bf16_t f2bf(float f) {
  union { float f; unsigned u; } in;
  in.f = f;
  unsigned u = in.u;
  unsigned r = (u + 0x7FFFu + ((u >> 16) & 1u)) >> 16;  // round-to-nearest-even
  union { unsigned short s; bf16_t b; } out;
  out.s = (unsigned short)r;
  return out.b;
}

// ---------------------------------------------------------------------------
// Generic WMMA GEMM:  C(MxN) = alpha * A(MxK,bf16) * B(NxK,bf16)^T
//                              [+ C_old if beta] [+ bias[col]]
// grid = (M/128, N/128), block = 256.
// ---------------------------------------------------------------------------
__global__ __launch_bounds__(256) void gemm_bt_bf16(
    const bf16_t* __restrict__ A, int lda,
    const bf16_t* __restrict__ B, int ldb,
    const float* __restrict__ bias,
    float* Cf, bf16_t* __restrict__ Cbf, int ldc,
    int K, float alpha, int beta)
{
  __shared__ __align__(16) bf16_t As[2][128][40];   // 32 K + 8 pad
  __shared__ __align__(16) bf16_t Bs[2][128][40];

  const int tid  = threadIdx.x;
  const int lane = tid & 31;
  const int wave = tid >> 5;
  const int wm   = (wave & 3) * 32;   // wave M offset within block tile
  const int wn   = (wave >> 2) * 64;  // wave N offset within block tile
  const int lr   = lane & 15;
  const int lh   = lane >> 4;

  const int blockM = blockIdx.x * 128;
  const int blockN = blockIdx.y * 128;

  // tile-fill assignment: 8-element (16B) chunks; each thread moves 2 A + 2 B
  const int r0 = tid >> 2;            // 0..63
  const int r1 = r0 + 64;             // 64..127
  const int c0 = (tid & 3) * 8;       // 0,8,16,24

  const bf16_t* Ab = A + (size_t)blockM * lda;
  const bf16_t* Bb = B + (size_t)blockN * ldb;

  v8f acc[2][4] = {};
  const int Kt = K >> 5;

#if GEMM_ASYNC
  // ---- async global->LDS, double-buffered ----
  #define ISSUE_TILE(kt_, buf_)                                                \
    do {                                                                       \
      const int k0_ = (kt_) << 5;                                              \
      __builtin_amdgcn_global_load_async_to_lds_b128(                          \
          (gv4i_p)(Ab + (size_t)r0 * lda + k0_ + c0),                          \
          (lv4i_p)&As[buf_][r0][c0], 0, 0);                                    \
      __builtin_amdgcn_global_load_async_to_lds_b128(                          \
          (gv4i_p)(Ab + (size_t)r1 * lda + k0_ + c0),                          \
          (lv4i_p)&As[buf_][r1][c0], 0, 0);                                    \
      __builtin_amdgcn_global_load_async_to_lds_b128(                          \
          (gv4i_p)(Bb + (size_t)r0 * ldb + k0_ + c0),                          \
          (lv4i_p)&Bs[buf_][r0][c0], 0, 0);                                    \
      __builtin_amdgcn_global_load_async_to_lds_b128(                          \
          (gv4i_p)(Bb + (size_t)r1 * ldb + k0_ + c0),                          \
          (lv4i_p)&Bs[buf_][r1][c0], 0, 0);                                    \
    } while (0)

  ISSUE_TILE(0, 0);
  for (int kt = 0; kt < Kt; ++kt) {
    const int cur = kt & 1;
    if (kt + 1 < Kt) {
      ISSUE_TILE(kt + 1, cur ^ 1);               // keep next K tile in flight
      __builtin_amdgcn_s_wait_asynccnt(4);       // tile kt complete, kt+1 pending
    } else {
      __builtin_amdgcn_s_wait_asynccnt(0);
    }
    if (kt + 2 < Kt) {                           // L2 prefetch two tiles ahead
      __builtin_prefetch(Ab + (size_t)r0 * lda + ((kt + 2) << 5) + c0, 0, 1);
    }
    __syncthreads();

    // A fragment: lane -> row m = lr; K(e) = (e&7) + 8*lh + 16*(e>>3)
    FragBF fa[2], fb[4];
    #pragma unroll
    for (int mi = 0; mi < 2; ++mi) {
      const int r = wm + mi * 16 + lr;
      fa[mi].h[0] = *(const v8bf*)&As[cur][r][lh * 8];
      fa[mi].h[1] = *(const v8bf*)&As[cur][r][16 + lh * 8];
    }
    // B fragment: lane -> col n = lr; K(e) = (e&7) + 8*lh + 16*(e>>3)
    #pragma unroll
    for (int ni = 0; ni < 4; ++ni) {
      const int r = wn + ni * 16 + lr;
      fb[ni].h[0] = *(const v8bf*)&Bs[cur][r][lh * 8];
      fb[ni].h[1] = *(const v8bf*)&Bs[cur][r][16 + lh * 8];
    }
    #pragma unroll
    for (int mi = 0; mi < 2; ++mi)
      #pragma unroll
      for (int ni = 0; ni < 4; ++ni)
        acc[mi][ni] = __builtin_amdgcn_wmma_f32_16x16x32_bf16(
            false, fa[mi].v, false, fb[ni].v, (short)0, acc[mi][ni], false, false);

    __syncthreads();   // protect buffer `cur` before next iteration overwrites it
  }
  #undef ISSUE_TILE
#else
  // ---- fallback: synchronous global->VGPR->LDS, single buffer ----
  for (int kt = 0; kt < Kt; ++kt) {
    const int k0 = kt << 5;
    v8bf ga0 = *(const v8bf*)(Ab + (size_t)r0 * lda + k0 + c0);
    v8bf ga1 = *(const v8bf*)(Ab + (size_t)r1 * lda + k0 + c0);
    v8bf gb0 = *(const v8bf*)(Bb + (size_t)r0 * ldb + k0 + c0);
    v8bf gb1 = *(const v8bf*)(Bb + (size_t)r1 * ldb + k0 + c0);
    if (kt + 1 < Kt) {
      __builtin_prefetch(Ab + (size_t)r0 * lda + k0 + 32 + c0, 0, 1);
    }
    __syncthreads();
    *(v8bf*)&As[0][r0][c0] = ga0;
    *(v8bf*)&As[0][r1][c0] = ga1;
    *(v8bf*)&Bs[0][r0][c0] = gb0;
    *(v8bf*)&Bs[0][r1][c0] = gb1;
    __syncthreads();

    FragBF fa[2], fb[4];
    #pragma unroll
    for (int mi = 0; mi < 2; ++mi) {
      const int r = wm + mi * 16 + lr;
      fa[mi].h[0] = *(const v8bf*)&As[0][r][lh * 8];
      fa[mi].h[1] = *(const v8bf*)&As[0][r][16 + lh * 8];
    }
    #pragma unroll
    for (int ni = 0; ni < 4; ++ni) {
      const int r = wn + ni * 16 + lr;
      fb[ni].h[0] = *(const v8bf*)&Bs[0][r][lh * 8];
      fb[ni].h[1] = *(const v8bf*)&Bs[0][r][16 + lh * 8];
    }
    #pragma unroll
    for (int mi = 0; mi < 2; ++mi)
      #pragma unroll
      for (int ni = 0; ni < 4; ++ni)
        acc[mi][ni] = __builtin_amdgcn_wmma_f32_16x16x32_bf16(
            false, fa[mi].v, false, fb[ni].v, (short)0, acc[mi][ni], false, false);
  }
#endif

  // Epilogue. C layout: VGPR r -> M = r + 8*lh (+subtile), N = lr (+subtile).
  #pragma unroll
  for (int mi = 0; mi < 2; ++mi) {
    #pragma unroll
    for (int ni = 0; ni < 4; ++ni) {
      const int col = blockN + wn + ni * 16 + lr;
      const float bv = bias ? bias[col] : 0.0f;
      const int rbase = blockM + wm + mi * 16 + lh * 8;
      #pragma unroll
      for (int r = 0; r < 8; ++r) {
        const size_t idx = (size_t)(rbase + r) * ldc + col;
        float v = alpha * acc[mi][ni][r] + bv;
        if (beta) v += Cf[idx];
        if (Cf)  Cf[idx]  = v;
        if (Cbf) Cbf[idx] = f2bf(v);
      }
    }
  }
}

// ---------------------------------------------------------------------------
// Row LayerNorm + exact GELU. One block per row.
// ---------------------------------------------------------------------------
__global__ __launch_bounds__(256) void ln_gelu_k(
    const float* __restrict__ X, const float* __restrict__ g,
    const float* __restrict__ be, float* __restrict__ outf,
    bf16_t* __restrict__ outb, int cols)
{
  __shared__ float s1[256];
  __shared__ float s2[256];
  const int row = blockIdx.x;
  const int tid = threadIdx.x;
  const float* xr = X + (size_t)row * cols;
  float sum = 0.f, sq = 0.f;
  for (int c = tid; c < cols; c += 256) { float v = xr[c]; sum += v; sq += v * v; }
  s1[tid] = sum; s2[tid] = sq;
  __syncthreads();
  for (int s = 128; s > 0; s >>= 1) {
    if (tid < s) { s1[tid] += s1[tid + s]; s2[tid] += s2[tid + s]; }
    __syncthreads();
  }
  const float mean = s1[0] / (float)cols;
  const float var  = s2[0] / (float)cols - mean * mean;
  const float rstd = rsqrtf(var + 1e-5f);
  for (int c = tid; c < cols; c += 256) {
    float v  = (xr[c] - mean) * rstd * g[c] + be[c];
    float ge = 0.5f * v * (1.0f + erff(v * 0.70710678118654752f));
    const size_t idx = (size_t)row * cols + c;
    if (outf) outf[idx] = ge;
    if (outb) outb[idx] = f2bf(ge);
  }
}

// ---------------------------------------------------------------------------
// Complex-magnitude attention. One 64-thread block per (b, h).
// N=4 slots, HD=64. q/k/v layout: (B, N, D=H*HD) f32. Output bf16 (B,N,D).
// ---------------------------------------------------------------------------
__global__ __launch_bounds__(64) void attention_k(
    const float* __restrict__ qr, const float* __restrict__ qi,
    const float* __restrict__ kr, const float* __restrict__ ki,
    const float* __restrict__ vr, const float* __restrict__ vi,
    bf16_t* __restrict__ outr, bf16_t* __restrict__ outi)
{
  __shared__ float red[64 * 32];
  __shared__ float rsum[32];
  const int b   = blockIdx.x >> 3;   // H = 8
  const int hh  = blockIdx.x & 7;
  const int tid = threadIdx.x;       // hd index 0..63
  float QR[4], QI[4], KR[4], KI[4], VR[4], VI[4];
  #pragma unroll
  for (int n = 0; n < 4; ++n) {
    const size_t idx = ((size_t)(b * 4 + n)) * 512 + hh * 64 + tid;
    QR[n] = qr[idx]; QI[n] = qi[idx];
    KR[n] = kr[idx]; KI[n] = ki[idx];
    VR[n] = vr[idx]; VI[n] = vi[idx];
  }
  #pragma unroll
  for (int n = 0; n < 4; ++n)
    #pragma unroll
    for (int m = 0; m < 4; ++m) {
      red[tid * 32 + (n * 4 + m)]      = QR[n] * KR[m] + QI[n] * KI[m];  // ar part
      red[tid * 32 + 16 + (n * 4 + m)] = QI[n] * KR[m] - QR[n] * KI[m];  // ai part
    }
  __syncthreads();
  if (tid < 32) {
    float s = 0.f;
    for (int t = 0; t < 64; ++t) s += red[t * 32 + tid];
    rsum[tid] = s;
  }
  __syncthreads();
  float w[4][4];
  #pragma unroll
  for (int n = 0; n < 4; ++n) {
    float mg[4], mx = -1e30f;
    #pragma unroll
    for (int m = 0; m < 4; ++m) {
      const float ar = rsum[n * 4 + m];
      const float ai = rsum[16 + n * 4 + m];
      mg[m] = sqrtf(ar * ar + ai * ai + 1e-8f) * 0.125f;  // HD^-0.5 = 1/8
      mx = fmaxf(mx, mg[m]);
    }
    float s = 0.f;
    #pragma unroll
    for (int m = 0; m < 4; ++m) { w[n][m] = expf(mg[m] - mx); s += w[n][m]; }
    const float inv = 1.0f / s;
    #pragma unroll
    for (int m = 0; m < 4; ++m) w[n][m] *= inv;
  }
  #pragma unroll
  for (int n = 0; n < 4; ++n) {
    float orr = 0.f, oii = 0.f;
    #pragma unroll
    for (int m = 0; m < 4; ++m) { orr += w[n][m] * VR[m]; oii += w[n][m] * VI[m]; }
    const size_t idx = ((size_t)(b * 4 + n)) * 512 + hh * 64 + tid;
    outr[idx] = f2bf(orr);
    outi[idx] = f2bf(oii);
  }
}

// ---------------------------------------------------------------------------
// Phase interference + superposition + L2 normalize. One 512-thread block / b.
// cos(atan2(y,x)) == x / sqrt(x^2+y^2)  (==1 at origin, matching atan2(0,0)=0)
// ---------------------------------------------------------------------------
__global__ __launch_bounds__(512) void superpose_k(
    const float* __restrict__ R, const float* __restrict__ I,
    const float* __restrict__ strength,
    bf16_t* __restrict__ supr, bf16_t* __restrict__ supi)
{
  __shared__ float sS[16];
  __shared__ float red[512];
  const int b = blockIdx.x;
  const int d = threadIdx.x;
  if (d < 16) sS[d] = strength[b * 16 + d];
  float R4[4], I4[4];
  #pragma unroll
  for (int n = 0; n < 4; ++n) {
    const size_t idx = ((size_t)(b * 4 + n)) * 512 + d;
    R4[n] = R[idx]; I4[n] = I[idx];
  }
  __syncthreads();
  float sr = 0.f, si = 0.f;
  #pragma unroll
  for (int i = 0; i < 4; ++i)
    #pragma unroll
    for (int j = 0; j < 4; ++j) {
      const float x = R4[i] * R4[j] + I4[i] * I4[j];
      const float y = I4[i] * R4[j] - R4[i] * I4[j];
      const float h = sqrtf(x * x + y * y);
      const float c = (h > 0.f) ? (x / h) : 1.0f;
      const float f = c * sS[i * 4 + j];
      sr += f * R4[j];
      si += f * I4[j];
    }
  red[d] = sr * sr + si * si;
  __syncthreads();
  for (int s = 256; s > 0; s >>= 1) {
    if (d < s) red[d] += red[d + s];
    __syncthreads();
  }
  const float rn = 1.0f / sqrtf(red[0] + 1e-8f);
  supr[(size_t)b * 512 + d] = f2bf(sr * rn);
  supi[(size_t)b * 512 + d] = f2bf(si * rn);
}

// ---------------------------------------------------------------------------
__global__ __launch_bounds__(256) void amp_k(
    const float* __restrict__ mr, const float* __restrict__ mi,
    bf16_t* __restrict__ out, size_t n)
{
  size_t i = (size_t)blockIdx.x * blockDim.x + threadIdx.x;
  const size_t stride = (size_t)gridDim.x * blockDim.x;
  for (; i < n; i += stride)
    out[i] = f2bf(sqrtf(mr[i] * mr[i] + mi[i] * mi[i]));
}

__global__ __launch_bounds__(256) void cvt_bf16_k(
    const float* __restrict__ s, bf16_t* __restrict__ d, size_t n)
{
  size_t i = (size_t)blockIdx.x * blockDim.x + threadIdx.x;
  const size_t stride = (size_t)gridDim.x * blockDim.x;
  for (; i < n; i += stride) d[i] = f2bf(s[i]);
}

// ---------------------------------------------------------------------------
extern "C" void kernel_launch(void* const* d_in, const int* in_sizes, int n_in,
                              void* d_out, int out_size, void* d_ws, size_t ws_size,
                              hipStream_t stream)
{
  (void)in_sizes; (void)n_in; (void)out_size; (void)ws_size;
  constexpr int Bz = 4096, INz = 2048, HIDz = 2048, Dz = 512, Nz = 4, NDz = 2048, OUTz = 512, Hz = 8;

  const float* x        = (const float*)d_in[0];
  const float* strength = (const float*)d_in[1];
  const float* er_W1 = (const float*)d_in[2];
  const float* er_b1 = (const float*)d_in[3];
  const float* er_g1 = (const float*)d_in[4];
  const float* er_be1= (const float*)d_in[5];
  const float* er_W2 = (const float*)d_in[6];
  const float* er_b2 = (const float*)d_in[7];
  const float* ei_W1 = (const float*)d_in[8];
  const float* ei_b1 = (const float*)d_in[9];
  const float* ei_g1 = (const float*)d_in[10];
  const float* ei_be1= (const float*)d_in[11];
  const float* ei_W2 = (const float*)d_in[12];
  const float* ei_b2 = (const float*)d_in[13];
  const float* Wq_r = (const float*)d_in[14]; const float* bq_r = (const float*)d_in[15];
  const float* Wq_i = (const float*)d_in[16]; const float* bq_i = (const float*)d_in[17];
  const float* Wk_r = (const float*)d_in[18]; const float* bk_r = (const float*)d_in[19];
  const float* Wk_i = (const float*)d_in[20]; const float* bk_i = (const float*)d_in[21];
  const float* Wv_r = (const float*)d_in[22]; const float* bv_r = (const float*)d_in[23];
  const float* Wv_i = (const float*)d_in[24]; const float* bv_i = (const float*)d_in[25];
  const float* Wo_r = (const float*)d_in[26]; const float* bo_r = (const float*)d_in[27];
  const float* Wo_i = (const float*)d_in[28]; const float* bo_i = (const float*)d_in[29];
  const float* int_Wr = (const float*)d_in[30];
  const float* int_Wi = (const float*)d_in[31];
  const float* meas_Mr = (const float*)d_in[32];
  const float* meas_Mi = (const float*)d_in[33];
  const float* post_W = (const float*)d_in[34];
  const float* post_b = (const float*)d_in[35];
  const float* post_g = (const float*)d_in[36];
  const float* post_be= (const float*)d_in[37];

  size_t off = 0;
  auto alloc = [&](size_t bytes) -> void* {
    void* p = (char*)d_ws + off;
    off += (bytes + 255) & ~(size_t)255;
    return p;
  };

  // bf16 buffers (persistent across pipeline)
  bf16_t* xb    = (bf16_t*)alloc((size_t)Bz * INz * 2);
  bf16_t* wb1e  = (bf16_t*)alloc((size_t)HIDz * INz * 2);
  bf16_t* wb1i  = (bf16_t*)alloc((size_t)HIDz * INz * 2);
  bf16_t* wb2e  = (bf16_t*)alloc((size_t)NDz * HIDz * 2);
  bf16_t* wb2i  = (bf16_t*)alloc((size_t)NDz * HIDz * 2);
  bf16_t* wqr   = (bf16_t*)alloc((size_t)Dz * Dz * 2);
  bf16_t* wqi   = (bf16_t*)alloc((size_t)Dz * Dz * 2);
  bf16_t* wkr   = (bf16_t*)alloc((size_t)Dz * Dz * 2);
  bf16_t* wki   = (bf16_t*)alloc((size_t)Dz * Dz * 2);
  bf16_t* wvr   = (bf16_t*)alloc((size_t)Dz * Dz * 2);
  bf16_t* wvi   = (bf16_t*)alloc((size_t)Dz * Dz * 2);
  bf16_t* wor   = (bf16_t*)alloc((size_t)Dz * Dz * 2);
  bf16_t* woi   = (bf16_t*)alloc((size_t)Dz * Dz * 2);
  bf16_t* wintr = (bf16_t*)alloc((size_t)Nz * Dz * Dz * 2);
  bf16_t* winti = (bf16_t*)alloc((size_t)Nz * Dz * Dz * 2);
  bf16_t* wmrb  = (bf16_t*)alloc((size_t)OUTz * Dz * 2);
  bf16_t* wmib  = (bf16_t*)alloc((size_t)OUTz * Dz * 2);
  bf16_t* wpost = (bf16_t*)alloc((size_t)OUTz * OUTz * 2);

  // activations
  float*  h    = (float*) alloc((size_t)Bz * HIDz * 4);
  bf16_t* hb   = (bf16_t*)alloc((size_t)Bz * HIDz * 2);
  bf16_t* srb  = (bf16_t*)alloc((size_t)Bz * NDz * 2);
  bf16_t* sib  = (bf16_t*)alloc((size_t)Bz * NDz * 2);
  float*  qrF  = (float*) alloc((size_t)Bz * NDz * 4);
  float*  qiF  = (float*) alloc((size_t)Bz * NDz * 4);
  float*  krF  = (float*) alloc((size_t)Bz * NDz * 4);
  float*  kiF  = (float*) alloc((size_t)Bz * NDz * 4);
  float*  vrF  = (float*) alloc((size_t)Bz * NDz * 4);
  float*  viF  = (float*) alloc((size_t)Bz * NDz * 4);

  // aliases (lifetimes verified to not overlap)
  bf16_t* outrb  = srb;             // attention output re-uses srb/sib
  bf16_t* outib  = sib;
  bf16_t* attnrb = (bf16_t*)h;      // Wo projection output re-uses h/hb
  bf16_t* attnib = hb;
  float*  Rb     = qrF;             // interference output re-uses qr/qi
  float*  Ib     = qiF;
  bf16_t* suprb  = (bf16_t*)krF;    // superposition output re-uses kr/ki
  bf16_t* supib  = (bf16_t*)kiF;
  float*  mrF    = vrF;             // measurement re-uses vr/vi
  float*  miF    = viF;
  bf16_t* ampb   = outrb;           // amplitude re-uses srb region (dead)
  float*  postF  = Rb;              // post-linear re-uses R region (dead)

  auto cvt = [&](const float* s, bf16_t* d, size_t n) {
    int blocks = (int)((n + 2047) / 2048);
    if (blocks > 8192) blocks = 8192;
    cvt_bf16_k<<<blocks, 256, 0, stream>>>(s, d, n);
  };
  auto gemm = [&](const bf16_t* A, int lda, const bf16_t* B, int ldb,
                  const float* bias, float* Cf, bf16_t* Cbf, int ldc,
                  int M, int Nn, int K, float alpha, int beta) {
    dim3 grid(M / 128, Nn / 128);
    gemm_bt_bf16<<<grid, 256, 0, stream>>>(A, lda, B, ldb, bias, Cf, Cbf, ldc, K, alpha, beta);
  };

  // ---- fp32 -> bf16 conversions (inputs & weights) ----
  cvt(x,       xb,    (size_t)Bz * INz);
  cvt(er_W1,   wb1e,  (size_t)HIDz * INz);
  cvt(ei_W1,   wb1i,  (size_t)HIDz * INz);
  cvt(er_W2,   wb2e,  (size_t)NDz * HIDz);
  cvt(ei_W2,   wb2i,  (size_t)NDz * HIDz);
  cvt(Wq_r, wqr, (size_t)Dz * Dz); cvt(Wq_i, wqi, (size_t)Dz * Dz);
  cvt(Wk_r, wkr, (size_t)Dz * Dz); cvt(Wk_i, wki, (size_t)Dz * Dz);
  cvt(Wv_r, wvr, (size_t)Dz * Dz); cvt(Wv_i, wvi, (size_t)Dz * Dz);
  cvt(Wo_r, wor, (size_t)Dz * Dz); cvt(Wo_i, woi, (size_t)Dz * Dz);
  cvt(int_Wr,  wintr, (size_t)Nz * Dz * Dz);
  cvt(int_Wi,  winti, (size_t)Nz * Dz * Dz);
  cvt(meas_Mr, wmrb,  (size_t)OUTz * Dz);
  cvt(meas_Mi, wmib,  (size_t)OUTz * Dz);
  cvt(post_W,  wpost, (size_t)OUTz * OUTz);

  // ---- encoder er: h = LN_GELU(x @ W1^T + b1); sr = h @ W2^T + b2 ----
  gemm(xb, INz, wb1e, INz, er_b1, h, nullptr, HIDz, Bz, HIDz, INz, 1.0f, 0);
  ln_gelu_k<<<Bz, 256, 0, stream>>>(h, er_g1, er_be1, nullptr, hb, HIDz);
  gemm(hb, HIDz, wb2e, HIDz, er_b2, nullptr, srb, NDz, Bz, NDz, HIDz, 1.0f, 0);

  // ---- encoder ei ----
  gemm(xb, INz, wb1i, INz, ei_b1, h, nullptr, HIDz, Bz, HIDz, INz, 1.0f, 0);
  ln_gelu_k<<<Bz, 256, 0, stream>>>(h, ei_g1, ei_be1, nullptr, hb, HIDz);
  gemm(hb, HIDz, wb2i, HIDz, ei_b2, nullptr, sib, NDz, Bz, NDz, HIDz, 1.0f, 0);

  // ---- QKV projections on (B*N, D) ----
  const int BN = Bz * Nz;
  gemm(srb, Dz, wqr, Dz, bq_r, qrF, nullptr, Dz, BN, Dz, Dz, 1.0f, 0);
  gemm(sib, Dz, wqi, Dz, bq_i, qiF, nullptr, Dz, BN, Dz, Dz, 1.0f, 0);
  gemm(srb, Dz, wkr, Dz, bk_r, krF, nullptr, Dz, BN, Dz, Dz, 1.0f, 0);
  gemm(sib, Dz, wki, Dz, bk_i, kiF, nullptr, Dz, BN, Dz, Dz, 1.0f, 0);
  gemm(srb, Dz, wvr, Dz, bv_r, vrF, nullptr, Dz, BN, Dz, Dz, 1.0f, 0);
  gemm(sib, Dz, wvi, Dz, bv_i, viF, nullptr, Dz, BN, Dz, Dz, 1.0f, 0);

  // ---- attention (per (b, h) block) ----
  attention_k<<<Bz * Hz, 64, 0, stream>>>(qrF, qiF, krF, kiF, vrF, viF, outrb, outib);

  // ---- output projections ----
  gemm(outrb, Dz, wor, Dz, bo_r, nullptr, attnrb, Dz, BN, Dz, Dz, 1.0f, 0);
  gemm(outib, Dz, woi, Dz, bo_i, nullptr, attnib, Dz, BN, Dz, Dz, 1.0f, 0);

  // ---- interference: per-slot D x D complex matmul ----
  for (int n = 0; n < Nz; ++n) {
    const bf16_t* ar = attnrb + (size_t)n * Dz;
    const bf16_t* ai = attnib + (size_t)n * Dz;
    const bf16_t* wr = wintr + (size_t)n * Dz * Dz;
    const bf16_t* wi = winti + (size_t)n * Dz * Dz;
    float* Rn = Rb + (size_t)n * Dz;
    float* In = Ib + (size_t)n * Dz;
    gemm(ar, NDz, wr, Dz, nullptr, Rn, nullptr, NDz, Bz, Dz, Dz,  1.0f, 0);
    gemm(ai, NDz, wi, Dz, nullptr, Rn, nullptr, NDz, Bz, Dz, Dz, -1.0f, 1);
    gemm(ar, NDz, wi, Dz, nullptr, In, nullptr, NDz, Bz, Dz, Dz,  1.0f, 0);
    gemm(ai, NDz, wr, Dz, nullptr, In, nullptr, NDz, Bz, Dz, Dz,  1.0f, 1);
  }

  // ---- phase superposition + normalize ----
  superpose_k<<<Bz, 512, 0, stream>>>(Rb, Ib, strength, suprb, supib);

  // ---- complex measurement ----
  gemm(suprb, Dz, wmrb, Dz, nullptr, mrF, nullptr, OUTz, Bz, OUTz, Dz,  1.0f, 0);
  gemm(supib, Dz, wmib, Dz, nullptr, mrF, nullptr, OUTz, Bz, OUTz, Dz, -1.0f, 1);
  gemm(suprb, Dz, wmib, Dz, nullptr, miF, nullptr, OUTz, Bz, OUTz, Dz,  1.0f, 0);
  gemm(supib, Dz, wmrb, Dz, nullptr, miF, nullptr, OUTz, Bz, OUTz, Dz,  1.0f, 1);

  // ---- amplitude -> bf16 ----
  amp_k<<<2048, 256, 0, stream>>>(mrF, miF, ampb, (size_t)Bz * OUTz);

  // ---- post linear + LN + GELU -> d_out (f32) ----
  gemm(ampb, OUTz, wpost, OUTz, post_b, postF, nullptr, OUTz, Bz, OUTz, OUTz, 1.0f, 0);
  ln_gelu_k<<<Bz, 256, 0, stream>>>(postF, post_g, post_be, (float*)d_out, nullptr, OUTz);
}
